// GPT_38792144617670
// MI455X (gfx1250) — compile-verified, matrix operand order checked by reading, not run on hardware
//
#include <hip/hip_runtime.h>
#include <hip/hip_bf16.h>

// ---------------- model constants ----------------
#define VV 32000
#define DD 1024
#define TT 1024
#define HH 16
#define LL 8
#define BB 2
#define FFF (4 * DD)
#define HDD (DD / HH)
#define EPSF 1e-5f

typedef __attribute__((ext_vector_type(16))) _Float16 v16h;
typedef __attribute__((ext_vector_type(8)))  _Float16 v8h;
typedef __attribute__((ext_vector_type(2)))  _Float16 h2;
typedef __attribute__((ext_vector_type(8)))  float    v8f;

// ---------------- embedding ----------------
__global__ __launch_bounds__(256) void embed_kernel(
    const int* __restrict__ idx, const float* __restrict__ tok,
    const float* __restrict__ pos, float* __restrict__ x, long long n)
{
    long long i = (long long)blockIdx.x * blockDim.x + threadIdx.x;
    if (i >= n) return;
    int d = (int)(i % DD);
    long long bt = i / DD;
    int t = (int)(bt % TT);
    int tokid = idx[bt];
    x[i] = tok[(long long)tokid * DD + d] + pos[(long long)t * DD + d];
}

// ---------------- layernorm (one block per row) ----------------
__global__ __launch_bounds__(256) void ln_kernel(
    const float* __restrict__ x, const float* __restrict__ s,
    const float* __restrict__ b, float* __restrict__ y, int n_rows)
{
    __shared__ float red[256];
    int row = blockIdx.x;
    if (row >= n_rows) return;
    const float* xr = x + (long long)row * DD;
    float* yr = y + (long long)row * DD;
    int tid = threadIdx.x;

    float acc = 0.f;
    for (int i = tid; i < DD; i += 256) acc += xr[i];
    red[tid] = acc; __syncthreads();
    for (int off = 128; off > 0; off >>= 1) {
        if (tid < off) red[tid] += red[tid + off];
        __syncthreads();
    }
    float mu = red[0] / (float)DD;
    __syncthreads();

    acc = 0.f;
    for (int i = tid; i < DD; i += 256) { float d = xr[i] - mu; acc += d * d; }
    red[tid] = acc; __syncthreads();
    for (int off = 128; off > 0; off >>= 1) {
        if (tid < off) red[tid] += red[tid + off];
        __syncthreads();
    }
    float rstd = rsqrtf(red[0] / (float)DD + EPSF);

    for (int i = tid; i < DD; i += 256)
        yr[i] = (xr[i] - mu) * rstd * s[i] + b[i];
}

// ---------------- causal softmax over scores[z][q][0..T) ----------------
__global__ __launch_bounds__(256) void softmax_kernel(float* __restrict__ scores)
{
    __shared__ float red[256];
    int q = blockIdx.x;
    int z = blockIdx.y;
    float* row = scores + (long long)z * TT * TT + (long long)q * TT;
    int tid = threadIdx.x;
    int nvalid = q + 1;

    float mx = -3.4e38f;
    for (int i = tid; i < nvalid; i += 256) mx = fmaxf(mx, row[i]);
    red[tid] = mx; __syncthreads();
    for (int off = 128; off > 0; off >>= 1) {
        if (tid < off) red[tid] = fmaxf(red[tid], red[tid + off]);
        __syncthreads();
    }
    mx = red[0]; __syncthreads();

    float sum = 0.f;
    for (int i = tid; i < nvalid; i += 256) {
        float e = __expf(row[i] - mx);
        row[i] = e;
        sum += e;
    }
    red[tid] = sum; __syncthreads();
    for (int off = 128; off > 0; off >>= 1) {
        if (tid < off) red[tid] += red[tid + off];
        __syncthreads();
    }
    float inv = 1.f / red[0];

    for (int i = tid; i < TT; i += 256)
        row[i] = (i < nvalid) ? row[i] * inv : 0.f;
}

// ---------------- WMMA GEMM ----------------
// C[m,n] = alpha * sum_k A[m,k]*W(k,n) (+bias[n]) (+residual[m,n]) (relu)
// REQUIRES: M % 128 == 0, N % 64 == 0, K % 64 == 0 (true for all call sites).
// Block: 256 threads (8 waves), tile 128(M) x 64(N), K-step 64.
// Wave w computes rows [w*16, w*16+16) x all 64 N: 8 wmma per K-step, A frag
// reused 4x. LDS tiles are f16 in [row][K] order so each fragment is two
// 16-byte ds loads. Software pipeline: next tile's global loads are issued
// into registers while the current tile is multiplied from LDS.
#define KSTEP 64
#define LDSP 72   // padded row stride in halves (144B: 16B-aligned, bank-spread)

template <int TRANSB>
__global__ __launch_bounds__(256) void wmma_gemm_kernel(
    const float* __restrict__ A, int lda, long long sAb, long long sAh,
    const float* __restrict__ W, int ldb, long long sBb, long long sBh,
    float* __restrict__ C, int ldc, long long sCb, long long sCh,
    const float* __restrict__ bias, const float* __restrict__ residual,
    int M, int N, int K, int Hh, float alpha, int relu)
{
    __shared__ _Float16 ldsA[128][LDSP];  // [m][k]
    __shared__ _Float16 ldsBt[64][LDSP];  // [n][k]

    int z = blockIdx.z;
    int bb = z / Hh;
    int hh = z - bb * Hh;
    const float* Ab = A + bb * sAb + hh * sAh;
    const float* Wb = W + bb * sBb + hh * sBh;
    float*       Cb = C + bb * sCb + hh * sCh;
    const float* Rb = residual ? (residual + bb * sCb + hh * sCh) : nullptr;

    const int n0 = blockIdx.x * 64;
    const int m0 = blockIdx.y * 128;

    const int tid  = threadIdx.x;
    const int lane = tid & 31;
    const int wv   = tid >> 5;        // 0..7 -> M strip
    const int l2   = lane & 15;
    const int hi   = lane >> 4;       // K-half select per ISA layout

    // staging registers (software pipeline)
    float2 ra[16];                    // A: 128x64 = 4096 float2 / 256 thr
    float2 rbT[8];                    // B (transB): 64x64 = 2048 float2 / 256 thr
    float  rbN[16];                   // B (normal): 4096 floats / 256 thr

    // per-thread staging coordinates
    const int arow = tid >> 3;        // with i: full (r,c2) computed below

    auto loadA = [&](int kt) {
        #pragma unroll
        for (int i = 0; i < 16; ++i) {
            int p = tid + i * 256;
            int r = p >> 5, c2 = (p & 31) * 2;
            ra[i] = *(const float2*)(Ab + (long long)(m0 + r) * lda + kt + c2);
        }
    };
    auto loadB = [&](int kt) {
        if (TRANSB) {
            #pragma unroll
            for (int i = 0; i < 8; ++i) {
                int p = tid + i * 256;
                int nn = p >> 5, c2 = (p & 31) * 2;
                rbT[i] = *(const float2*)(Wb + (long long)(n0 + nn) * ldb + kt + c2);
            }
        } else {
            #pragma unroll
            for (int i = 0; i < 16; ++i) {
                int e = tid + i * 256;
                int k = e >> 6, nn = e & 63;
                rbN[i] = Wb[(long long)(kt + k) * ldb + n0 + nn];
            }
        }
    };
    auto storeTiles = [&]() {
        #pragma unroll
        for (int i = 0; i < 16; ++i) {
            int p = tid + i * 256;
            int r = p >> 5, c2 = (p & 31) * 2;
            h2 hv; hv.x = (_Float16)ra[i].x; hv.y = (_Float16)ra[i].y;
            *(h2*)&ldsA[r][c2] = hv;
        }
        if (TRANSB) {
            #pragma unroll
            for (int i = 0; i < 8; ++i) {
                int p = tid + i * 256;
                int nn = p >> 5, c2 = (p & 31) * 2;
                h2 hv; hv.x = (_Float16)rbT[i].x; hv.y = (_Float16)rbT[i].y;
                *(h2*)&ldsBt[nn][c2] = hv;
            }
        } else {
            #pragma unroll
            for (int i = 0; i < 16; ++i) {
                int e = tid + i * 256;
                int k = e >> 6, nn = e & 63;
                ldsBt[nn][k] = (_Float16)rbN[i];
            }
        }
    };
    (void)arow;

    v8f acc0 = {}, acc1 = {}, acc2 = {}, acc3 = {};

    loadA(0);
    loadB(0);

    for (int kt = 0; kt < K; kt += KSTEP) {
        storeTiles();
        __syncthreads();

        if (kt + KSTEP < K) {         // issue next tile's global loads now
            loadA(kt + KSTEP);
            loadB(kt + KSTEP);
        }

        // multiply current tile: 2 K-substeps of 32, 4 N-subtiles each
        #pragma unroll
        for (int ks = 0; ks < KSTEP; ks += 32) {
            v8h alo = *(const v8h*)&ldsA[wv * 16 + l2][ks + hi * 8];
            v8h ahi = *(const v8h*)&ldsA[wv * 16 + l2][ks + 16 + hi * 8];
            v16h av = __builtin_shufflevector(alo, ahi,
                        0,1,2,3,4,5,6,7,8,9,10,11,12,13,14,15);
            #pragma unroll
            for (int j = 0; j < 4; ++j) {
                v8h blo = *(const v8h*)&ldsBt[j * 16 + l2][ks + hi * 8];
                v8h bhi = *(const v8h*)&ldsBt[j * 16 + l2][ks + 16 + hi * 8];
                v16h bv = __builtin_shufflevector(blo, bhi,
                            0,1,2,3,4,5,6,7,8,9,10,11,12,13,14,15);
                if (j == 0) acc0 = __builtin_amdgcn_wmma_f32_16x16x32_f16(false, av, false, bv, (short)0, acc0, false, false);
                if (j == 1) acc1 = __builtin_amdgcn_wmma_f32_16x16x32_f16(false, av, false, bv, (short)0, acc1, false, false);
                if (j == 2) acc2 = __builtin_amdgcn_wmma_f32_16x16x32_f16(false, av, false, bv, (short)0, acc2, false, false);
                if (j == 3) acc3 = __builtin_amdgcn_wmma_f32_16x16x32_f16(false, av, false, bv, (short)0, acc3, false, false);
            }
        }
        __syncthreads();
    }

    // ---- epilogue: lane l2 owns column j*16+l2; rows r + hi*8 (VGPR r)
    #pragma unroll
    for (int j = 0; j < 4; ++j) {
        const v8f acc = (j == 0) ? acc0 : (j == 1) ? acc1 : (j == 2) ? acc2 : acc3;
        int n = n0 + j * 16 + l2;
        float bval = bias ? bias[n] : 0.f;
        #pragma unroll
        for (int r = 0; r < 8; ++r) {
            int m = m0 + wv * 16 + r + hi * 8;
            long long ci = (long long)m * ldc + n;
            float v = acc[r] * alpha + bval;
            if (Rb) v += Rb[ci];
            if (relu) v = fmaxf(v, 0.f);
            Cb[ci] = v;
        }
    }
}

// ---------------- host-side GEMM launcher ----------------
static void gemm(hipStream_t st,
                 const float* A, int lda, long long sAb, long long sAh,
                 const float* W, int ldb, long long sBb, long long sBh, int transB,
                 float* C, int ldc, long long sCb, long long sCh,
                 const float* bias, const float* residual,
                 int M, int N, int K, int Hh, int Z, float alpha, int relu)
{
    dim3 g(N / 64, M / 128, Z), blk(256);
    if (transB)
        wmma_gemm_kernel<1><<<g, blk, 0, st>>>(A, lda, sAb, sAh, W, ldb, sBb, sBh,
                                               C, ldc, sCb, sCh, bias, residual,
                                               M, N, K, Hh, alpha, relu);
    else
        wmma_gemm_kernel<0><<<g, blk, 0, st>>>(A, lda, sAb, sAh, W, ldb, sBb, sBh,
                                               C, ldc, sCb, sCh, bias, residual,
                                               M, N, K, Hh, alpha, relu);
}

extern "C" void kernel_launch(void* const* d_in, const int* in_sizes, int n_in,
                              void* d_out, int out_size, void* d_ws, size_t ws_size,
                              hipStream_t stream)
{
    (void)in_sizes; (void)n_in; (void)out_size; (void)ws_size;

    const int*   idx    = (const int*)  d_in[0];
    const float* tok    = (const float*)d_in[1];
    const float* pos    = (const float*)d_in[2];
    const float* ln1_s  = (const float*)d_in[3];
    const float* ln1_b  = (const float*)d_in[4];
    const float* wq     = (const float*)d_in[5];
    const float* wk     = (const float*)d_in[6];
    const float* wvp    = (const float*)d_in[7];
    const float* wo     = (const float*)d_in[8];
    const float* bo     = (const float*)d_in[9];
    const float* ln2_s  = (const float*)d_in[10];
    const float* ln2_b  = (const float*)d_in[11];
    const float* w1     = (const float*)d_in[12];
    const float* b1     = (const float*)d_in[13];
    const float* w2     = (const float*)d_in[14];
    const float* b2     = (const float*)d_in[15];
    const float* lnf_s  = (const float*)d_in[16];
    const float* lnf_b  = (const float*)d_in[17];
    const float* w_lm   = (const float*)d_in[18];
    const float* b_lm   = (const float*)d_in[19];
    float* out = (float*)d_out;

    const int M  = BB * TT;                  // 2048 rows
    const long long nAct = (long long)M * DD;

    // workspace layout (floats)
    float* ws   = (float*)d_ws;
    float* x    = ws;            ws += nAct;                       // [B*T, D]
    float* h    = ws;            ws += nAct;                       // LN output
    float* q    = ws;            ws += nAct;                       // [B,T,H,HD]
    float* kbuf = ws;            ws += nAct;
    float* vbuf = ws;            ws += nAct;
    float* attn = ws;            ws += nAct;                       // concat heads
    float* ffn1 = ws;            ws += (long long)M * FFF;         // [B*T, FF]
    float* scor = ws;            ws += (long long)BB * HH * TT * TT; // [B,H,T,T]

    // ---- embedding ----
    {
        long long n = nAct;
        dim3 g((unsigned)((n + 255) / 256)), blk(256);
        embed_kernel<<<g, blk, 0, stream>>>(idx, tok, pos, x, n);
    }

    const long long sBH_q  = (long long)TT * DD;   // per-batch stride in q/k/v/x
    const long long sH_q   = HDD;                  // per-head column offset
    const long long sBH_sc = (long long)HH * TT * TT;
    const long long sH_sc  = (long long)TT * TT;

    for (int l = 0; l < LL; ++l) {
        const float* wq_l = wq + (long long)l * DD * DD;
        const float* wk_l = wk + (long long)l * DD * DD;
        const float* wv_l = wvp + (long long)l * DD * DD;
        const float* wo_l = wo + (long long)l * DD * DD;
        const float* w1_l = w1 + (long long)l * DD * FFF;
        const float* w2_l = w2 + (long long)l * FFF * DD;

        // h = LN1(x)
        ln_kernel<<<dim3(M), dim3(256), 0, stream>>>(x, ln1_s + l * DD, ln1_b + l * DD, h, M);

        // q,k,v = h @ W  (M=2048, N=1024, K=1024)
        gemm(stream, h, DD, 0, 0, wq_l, DD, 0, 0, 0, q,    DD, 0, 0, nullptr, nullptr, M, DD, DD, 1, 1, 1.f, 0);
        gemm(stream, h, DD, 0, 0, wk_l, DD, 0, 0, 0, kbuf, DD, 0, 0, nullptr, nullptr, M, DD, DD, 1, 1, 1.f, 0);
        gemm(stream, h, DD, 0, 0, wv_l, DD, 0, 0, 0, vbuf, DD, 0, 0, nullptr, nullptr, M, DD, DD, 1, 1, 1.f, 0);

        // scores[z] = q_z @ k_z^T / sqrt(HD)   (M=T, N=T, K=HD, z=B*H)
        gemm(stream,
             q,    DD, sBH_q, sH_q,
             kbuf, DD, sBH_q, sH_q, /*transB=*/1,
             scor, TT, sBH_sc, sH_sc,
             nullptr, nullptr,
             TT, TT, HDD, HH, BB * HH, rsqrtf((float)HDD), 0);

        // causal softmax
        softmax_kernel<<<dim3(TT, BB * HH), dim3(256), 0, stream>>>(scor);

        // attn[z] = scores_z @ v_z   (M=T, N=HD, K=T)
        gemm(stream,
             scor, TT, sBH_sc, sH_sc,
             vbuf, DD, sBH_q, sH_q, /*transB=*/0,
             attn, DD, sBH_q, sH_q,
             nullptr, nullptr,
             TT, HDD, TT, HH, BB * HH, 1.f, 0);

        // x = attn @ wo + bo + x   (in place residual)
        gemm(stream, attn, DD, 0, 0, wo_l, DD, 0, 0, 0, x, DD, 0, 0,
             bo + l * DD, x, M, DD, DD, 1, 1, 1.f, 0);

        // h = LN2(x)
        ln_kernel<<<dim3(M), dim3(256), 0, stream>>>(x, ln2_s + l * DD, ln2_b + l * DD, h, M);

        // ffn1 = relu(h @ w1 + b1)   (N=FF)
        gemm(stream, h, DD, 0, 0, w1_l, FFF, 0, 0, 0, ffn1, FFF, 0, 0,
             b1 + l * FFF, nullptr, M, FFF, DD, 1, 1, 1.f, 1);

        // x = ffn1 @ w2 + b2 + x
        gemm(stream, ffn1, FFF, 0, 0, w2_l, DD, 0, 0, 0, x, DD, 0, 0,
             b2 + l * DD, x, M, DD, FFF, 1, 1, 1.f, 0);
    }

    // h = LN_f(x)
    ln_kernel<<<dim3(M), dim3(256), 0, stream>>>(x, lnf_s, lnf_b, h, M);

    // logits = h @ w_lm + b_lm  (M=2048, N=32000, K=1024)
    gemm(stream, h, DD, 0, 0, w_lm, VV, 0, 0, 0, out, VV, 0, 0,
         b_lm, nullptr, M, VV, DD, 1, 1, 1.f, 0);
}
